// PilotNet_38362647888353
// MI455X (gfx1250) — 8-bit matrix operand order repaired and checked by reading, compiled, not run on hardware
//
#include <hip/hip_runtime.h>
#include <hip/hip_bf16.h>


typedef __attribute__((ext_vector_type(8))) int v8i;
typedef __attribute__((ext_vector_type(4))) int v4i;
typedef __attribute__((ext_vector_type(4))) unsigned int v4u;

static __device__ __forceinline__ float dev_scale(const unsigned int* mb, int idx) {
    return __uint_as_float(mb[idx]) * (1.0f / 7.0f);
}

__global__ void init_maxbits(unsigned int* mb) {
    if (threadIdx.x < 16) mb[threadIdx.x] = 0u;
}

__global__ void absmax_kernel(const float* __restrict__ w, int n,
                              unsigned int* __restrict__ mb, int widx) {
    float m = 0.0f;
    for (int i = blockIdx.x * blockDim.x + threadIdx.x; i < n;
         i += gridDim.x * blockDim.x)
        m = fmaxf(m, fabsf(w[i]));
    for (int off = 16; off > 0; off >>= 1)
        m = fmaxf(m, __shfl_xor(m, off, 32));
    if ((threadIdx.x & 31) == 0)
        atomicMax(&mb[widx], __float_as_uint(m));
}

__global__ void pack_kernel(const float* __restrict__ w, signed char* __restrict__ out,
                            int Co, int Kreal, int Kpad, int mode, int p0, int p1, int p2,
                            const unsigned int* __restrict__ mb, int widx) {
    const int n = blockIdx.y;
    const float s = dev_scale(mb, widx);
    const float inv = 1.0f / s;
    for (int k = blockIdx.x * blockDim.x + threadIdx.x; k < Kpad;
         k += gridDim.x * blockDim.x) {
        signed char q = 0;
        if (n < Co && k < Kreal) {
            int src = k;
            if (mode == 1) {
                const int ci = k % p0;
                const int khkw = k / p0;
                src = ci * p1 + khkw;
            } else if (mode == 2) {
                const int c = k % p0;
                const int hw = k / p0;
                const int h = hw / p2;
                const int ww = hw - h * p2;
                src = (c * p1 + h) * p2 + ww;
            }
            float v = rintf(w[n * Kreal + src] * inv);
            v = fminf(fmaxf(v, -7.0f), 7.0f);
            q = (signed char)(int)v;
        }
        out[(long)n * Kpad + k] = q;
    }
}

__global__ void conv1_kernel(const float* __restrict__ x, const float* __restrict__ w,
                             const float* __restrict__ bias,
                             const unsigned int* __restrict__ mb,
                             const float* __restrict__ ascale_p,
                             unsigned char* __restrict__ out) {
    constexpr int HI = 240, WI = 320, CO = 24, HO = 118, WO = 158;
    __shared__ float wq[CO * 25];
    __shared__ float bq[CO];
    const float s = dev_scale(mb, 0);
    for (int i = threadIdx.x; i < CO * 25; i += blockDim.x) {
        float q = fminf(fmaxf(rintf(w[i] / s), -7.0f), 7.0f);
        wq[i] = q * s;
    }
    for (int i = threadIdx.x; i < CO; i += blockDim.x) bq[i] = bias[i];
    __syncthreads();
    const float ascale = *ascale_p;
    const float inv = 1.0f / ascale;
    constexpr int TOTAL = 128 * CO * HO * WO;
    const int idx = blockIdx.x * blockDim.x + threadIdx.x;
    if (idx >= TOTAL) return;
    const int co = idx % CO;
    int t = idx / CO;
    const int wo = t % WO; t /= WO;
    const int ho = t % HO;
    const int b  = t / HO;
    const float* xp = x + ((long)b * HI + ho * 2) * WI + wo * 2;
    const float* wv = &wq[co * 25];
    float y = bq[co];
#pragma unroll
    for (int kh = 0; kh < 5; ++kh)
#pragma unroll
        for (int kw = 0; kw < 5; ++kw)
            y = fmaf(xp[kh * WI + kw], wv[kh * 5 + kw], y);
    float q = fminf(fmaxf(rintf(fmaxf(y, 0.0f) * inv), 0.0f), 15.0f);
    out[idx] = (unsigned char)q;
}

template <int CI, int HI, int WI, int CO, int HO, int WO, int KH, int KW, int ST,
          int NPAD, int KPAD, int NTW, int WIDX, bool FC>
__global__ void __launch_bounds__(256)
qgemm_wmma(const unsigned char* __restrict__ in,
           const signed char* __restrict__ wp,
           const float* __restrict__ bias,
           const unsigned int* __restrict__ mb,
           const float* __restrict__ ascale_p,
           unsigned char* __restrict__ out) {
    constexpr int BATCH = 128;
    constexpr int M = BATCH * HO * WO;
    constexpr int KREAL = CI * KH * KW;
    constexpr int MT = M / 16;
    constexpr int NT = NPAD / 16;
    constexpr int KT = KPAD / 64;
    constexpr int NGRP = NT / NTW;
    static_assert(M % 16 == 0, "all layers have M % 16 == 0");
    static_assert(NT % NTW == 0, "n-tiles split evenly across waves");
    static_assert(FC || (CI % 4 == 0), "NHWC dword gather needs CI % 4 == 0");
    static_assert(FC || (KREAL % 4 == 0), "K padding is dword-granular");

    __shared__ int2 ktab[FC ? 16 : (KPAD / 4)];
    if constexpr (!FC) {
        for (int g = threadIdx.x; g < KPAD / 4; g += blockDim.x) {
            const int k = g * 4;
            int o = 0, msk = 0;
            if (k < KREAL) {
                const int khkw = k / CI;
                const int ci = k - khkw * CI;
                const int kh = khkw / KW;
                const int kw = khkw - kh * KW;
                o = (kh * WI + kw) * CI + ci;
                msk = -1;
            }
            ktab[g] = make_int2(o, msk);
        }
        __syncthreads();
    }

    const int wave = threadIdx.x >> 5;
    const int lane = threadIdx.x & 31;
    const int tile = blockIdx.x * 8 + wave;
    if (tile >= MT * NGRP) return;
    const int ng = tile / MT;
    const int mt = tile - ng * MT;
    const int nt_base = ng * NTW;
    const int half = lane >> 4;
    const int l15 = lane & 15;

    const int mrow = mt * 16 + l15;
    unsigned int abase = 0;
    if constexpr (!FC) {
        const int bb = mrow / (HO * WO);
        const int rm = mrow - bb * (HO * WO);
        const int aho = rm / WO;
        const int awo = rm - aho * WO;
        abase = (unsigned int)(((bb * HI + aho * ST) * WI + awo * ST) * CI);
    } else {
        abase = (unsigned int)(mrow * KPAD);
    }
    unsigned int boff[NTW];
    int ncols[NTW];
#pragma unroll
    for (int n = 0; n < NTW; ++n) {
        ncols[n] = (nt_base + n) * 16 + l15;
        boff[n] = (unsigned int)(ncols[n] * KPAD);
    }

    auto load_frag = [&](int kt, v8i& A, v8i* Bf) {
        const int k0 = kt * 64;
#pragma unroll
        for (int v = 0; v < 8; ++v) {
            const int kba = ((v & 1) << 2) | ((v & 2) << 3) | (half << 3) | ((v >> 2) << 5);
            unsigned int word;
            if constexpr (FC) {
                word = *reinterpret_cast<const unsigned int*>(
                    in + (abase + (unsigned int)(k0 + kba)));
            } else {
                const int2 om = ktab[(k0 + kba) >> 2];
                word = *reinterpret_cast<const unsigned int*>(
                           in + (abase + (unsigned int)om.x)) &
                       (unsigned int)om.y;
            }
            A[v] = (int)word;
        }
#pragma unroll
        for (int n = 0; n < NTW; ++n) {
#pragma unroll
            for (int v = 0; v < 8; ++v) {
                const int kbb = ((v & 1) << 2) | ((v & 2) << 3) | (half << 3) | ((v >> 2) << 5);
                Bf[n][v] = *reinterpret_cast<const int*>(
                    wp + (boff[n] + (unsigned int)(k0 + kbb)));
            }
            if (k0 + 64 < KPAD)
                __builtin_prefetch(wp + (boff[n] + (unsigned int)(k0 + 64)), 0, 1);
        }
    };

    const v8i vz = {0, 0, 0, 0, 0, 0, 0, 0};
    v8i acc[NTW];
#pragma unroll
    for (int n = 0; n < NTW; ++n) acc[n] = vz;
    v8i aA = vz, aB = vz;
    v8i BA[NTW], BB[NTW];
#pragma unroll
    for (int n = 0; n < NTW; ++n) { BA[n] = vz; BB[n] = vz; }

    load_frag(0, aA, BA);
    int kt = 0;
    while (kt + 1 < KT) {
        load_frag(kt + 1, aB, BB);
#pragma unroll
        for (int n = 0; n < NTW; ++n)
            acc[n] = __builtin_amdgcn_wmma_i32_16x16x64_iu8(false, aA, true, BA[n],
                                                            acc[n], false, false);
        ++kt;
        if (kt + 1 < KT)
            load_frag(kt + 1, aA, BA);
#pragma unroll
        for (int n = 0; n < NTW; ++n)
            acc[n] = __builtin_amdgcn_wmma_i32_16x16x64_iu8(false, aB, true, BB[n],
                                                            acc[n], false, false);
        ++kt;
    }
    if (kt < KT) {
#pragma unroll
        for (int n = 0; n < NTW; ++n)
            acc[n] = __builtin_amdgcn_wmma_i32_16x16x64_iu8(false, aA, true, BA[n],
                                                            acc[n], false, false);
    }

    const float ascale = *ascale_p;
    const float oscale = ascale * dev_scale(mb, WIDX);
    const float inv = 1.0f / ascale;
    float bv[NTW];
    bool cval[NTW];
#pragma unroll
    for (int n = 0; n < NTW; ++n) {
        cval[n] = (ncols[n] < CO);
        bv[n] = cval[n] ? bias[ncols[n]] : 0.0f;
    }

#pragma unroll
    for (int r = 0; r < 8; ++r) {
        const int mr = mt * 16 + r + (half << 3);
        long obase;
        if constexpr (FC) {
            obase = (long)mr * NPAD;
        } else {
            const int bb = mr / (HO * WO);
            const int rm = mr - bb * (HO * WO);
            const int ho = rm / WO;
            const int wo = rm - ho * WO;
            obase = (((long)bb * HO + ho) * WO + wo) * CO;
        }
#pragma unroll
        for (int n = 0; n < NTW; ++n) {
            const float y = (float)acc[n][r] * oscale + bv[n];
            const float q = fminf(fmaxf(rintf(fmaxf(y, 0.0f) * inv), 0.0f), 15.0f);
            if constexpr (FC) {
                out[obase + ncols[n]] = cval[n] ? (unsigned char)q : (unsigned char)0;
            } else if (cval[n]) {
                out[obase + ncols[n]] = (unsigned char)q;
            }
        }
    }
}

__global__ void tdm_warm_kernel(unsigned long long gaddr, unsigned char* sink) {
    __shared__ unsigned char stage[16384];
#if defined(__HIP_DEVICE_COMPILE__) && __has_builtin(__builtin_amdgcn_tensor_load_to_lds)
    v4u g0;
    g0[0] = 1u;
    g0[1] = 0u;
    g0[2] = (unsigned int)(gaddr & 0xFFFFFFFFu);
    g0[3] = (unsigned int)((gaddr >> 32) & 0x1FFFFFFu)
            | (2u << 30);
    v8i g1 = {0, 0, 0, 0, 0, 0, 0, 0};
    g1[1] = (16384u & 0xFFFFu) << 16;
    g1[2] = (16384u >> 16) & 0xFFFFu;
    g1[3] = (16384u & 0xFFFFu) << 16;
    v4i gz = {0, 0, 0, 0};
#if __clang_major__ >= 23
    v8i g5 = {0, 0, 0, 0, 0, 0, 0, 0};
    __builtin_amdgcn_tensor_load_to_lds(g0, g1, gz, gz, g5, 0);
#else
    __builtin_amdgcn_tensor_load_to_lds(g0, g1, gz, gz, 0);
#endif
    __builtin_amdgcn_s_wait_tensorcnt(0);
#endif
    if (threadIdx.x == 0) sink[0] = stage[0];
}

__global__ void final_kernel(const unsigned char* __restrict__ a,
                             const float* __restrict__ ow, const float* __restrict__ ob,
                             const unsigned int* __restrict__ mb,
                             const float* __restrict__ ascale_p,
                             float* __restrict__ out) {
    const int b = blockIdx.x * blockDim.x + threadIdx.x;
    if (b >= 128) return;
    const float s = dev_scale(mb, 8);
    const float ascale = *ascale_p;
    float y = ob[0];
#pragma unroll
    for (int i = 0; i < 10; ++i) {
        const float q = fminf(fmaxf(rintf(ow[i] / s), -7.0f), 7.0f) * s;
        y = fmaf((float)a[b * 16 + i] * ascale, q, y);
    }
    out[b] = y;
}

extern "C" void kernel_launch(void* const* d_in, const int* in_sizes, int n_in,
                              void* d_out, int out_size, void* d_ws, size_t ws_size,
                              hipStream_t stream) {
    (void)in_sizes; (void)n_in; (void)out_size; (void)ws_size;
    const float* x   = (const float*)d_in[0];
    const float* w1  = (const float*)d_in[1];
    const float* b1  = (const float*)d_in[2];
    const float* w2  = (const float*)d_in[3];
    const float* b2  = (const float*)d_in[4];
    const float* w3  = (const float*)d_in[5];
    const float* b3  = (const float*)d_in[6];
    const float* w4  = (const float*)d_in[7];
    const float* b4  = (const float*)d_in[8];
    const float* w5  = (const float*)d_in[9];
    const float* b5  = (const float*)d_in[10];
    const float* fw1 = (const float*)d_in[11];
    const float* fb1 = (const float*)d_in[12];
    const float* fw2 = (const float*)d_in[13];
    const float* fb2 = (const float*)d_in[14];
    const float* fw3 = (const float*)d_in[15];
    const float* fb3 = (const float*)d_in[16];
    const float* ow  = (const float*)d_in[17];
    const float* ob  = (const float*)d_in[18];
    const float* asc = (const float*)d_in[19];
    float* out = (float*)d_out;

    char* ws = (char*)d_ws;
    size_t off = 0;
    auto carve = [&](size_t bytes) {
        char* p = ws + off;
        off = (off + bytes + 255) & ~(size_t)255;
        return p;
    };
    unsigned int*  mb   = (unsigned int*)carve(64);
    unsigned char* sink = (unsigned char*)carve(256);
    signed char*   wp2  = (signed char*)carve(48 * 640);
    signed char*   wp3  = (signed char*)carve(48 * 960);
    signed char*   wp4  = (signed char*)carve(64 * 448);
    signed char*   wp5  = (signed char*)carve(64 * 576);
    signed char*   wf1  = (signed char*)carve((size_t)128 * 48576);
    signed char*   wf2  = (signed char*)carve(64 * 128);
    signed char*   wf3  = (signed char*)carve(16 * 64);
    unsigned char* bufA = (unsigned char*)carve(57274368);
    unsigned char* bufB = (unsigned char*)carve(20224512);

    init_maxbits<<<1, 32, 0, stream>>>(mb);

    auto absmax = [&](const float* w, int n, int idx) {
        int blocks = (n + 255) / 256;
        if (blocks > 1024) blocks = 1024;
        absmax_kernel<<<blocks, 256, 0, stream>>>(w, n, mb, idx);
    };
    absmax(w1, 600, 0);
    absmax(w2, 21600, 1);
    absmax(w3, 43200, 2);
    absmax(w4, 27648, 3);
    absmax(w5, 36864, 4);
    absmax(fw1, 4857600, 5);
    absmax(fw2, 5000, 6);
    absmax(fw3, 500, 7);
    absmax(ow, 10, 8);

    auto pack = [&](const float* w, signed char* o, int Co, int Kreal, int Npad,
                    int Kpad, int mode, int p0, int p1, int p2, int idx) {
        int gx = (Kpad + 255) / 256;
        if (gx > 64) gx = 64;
        dim3 g(gx, Npad);
        pack_kernel<<<g, 256, 0, stream>>>(w, o, Co, Kreal, Kpad, mode, p0, p1, p2,
                                           mb, idx);
    };
    pack(w2, wp2, 36, 600, 48, 640, 1, 24, 25, 0, 1);
    pack(w3, wp3, 48, 900, 48, 960, 1, 36, 25, 0, 2);
    pack(w4, wp4, 64, 432, 64, 448, 1, 48, 9, 0, 3);
    pack(w5, wp5, 64, 576, 64, 576, 1, 64, 9, 0, 4);
    pack(fw1, wf1, 100, 48576, 128, 48576, 2, 64, 23, 33, 5);
    pack(fw2, wf2, 50, 100, 64, 128, 0, 0, 0, 0, 6);
    pack(fw3, wf3, 10, 50, 16, 64, 0, 0, 0, 0, 7);

    conv1_kernel<<<(57274368 + 255) / 256, 256, 0, stream>>>(x, w1, b1, mb, asc, bufA);

    tdm_warm_kernel<<<1, 32, 0, stream>>>((unsigned long long)(uintptr_t)wf1, sink);

    auto blocks8 = [](long tiles) { return (int)((tiles + 7) / 8); };

    qgemm_wmma<24, 118, 158, 36, 57, 77, 5, 5, 2, 48, 640, 3, 1, false>
        <<<blocks8(35112), 256, 0, stream>>>(bufA, wp2, b2, mb, asc, bufB);
    qgemm_wmma<36, 57, 77, 48, 27, 37, 5, 5, 2, 48, 960, 3, 2, false>
        <<<blocks8(7992), 256, 0, stream>>>(bufB, wp3, b3, mb, asc, bufA);
    qgemm_wmma<48, 27, 37, 64, 25, 35, 3, 3, 1, 64, 448, 4, 3, false>
        <<<blocks8(7000), 256, 0, stream>>>(bufA, wp4, b4, mb, asc, bufB);
    qgemm_wmma<64, 25, 35, 64, 23, 33, 3, 3, 1, 64, 576, 4, 4, false>
        <<<blocks8(6072), 256, 0, stream>>>(bufB, wp5, b5, mb, asc, bufA);

    qgemm_wmma<48576, 1, 1, 100, 1, 1, 1, 1, 1, 128, 48576, 4, 5, true>
        <<<blocks8(16), 256, 0, stream>>>(bufA, wf1, fb1, mb, asc, bufB);
    qgemm_wmma<100, 1, 1, 50, 1, 1, 1, 1, 1, 64, 128, 4, 6, true>
        <<<blocks8(8), 256, 0, stream>>>(bufB, wf2, fb2, mb, asc, bufA);
    qgemm_wmma<50, 1, 1, 10, 1, 1, 1, 1, 1, 16, 64, 1, 7, true>
        <<<blocks8(8), 256, 0, stream>>>(bufA, wf3, fb3, mb, asc, bufB);

    final_kernel<<<1, 128, 0, stream>>>(bufB, ow, ob, mb, asc, out);
}
